// BitNetLayer_16054587753068
// MI455X (gfx1250) — 8-bit matrix operand order repaired and checked by reading, compiled, not run on hardware
//
#include <hip/hip_runtime.h>
#include <hip/hip_bf16.h>
#include <stdint.h>

typedef int v8i __attribute__((ext_vector_type(8)));

#define HID    2048
#define NTOK   8192
#define NLAYER 8
#define EPSF   1e-5f

__global__ void wabs_partial(const float* __restrict__ W, float* __restrict__ partials) {
    __shared__ float sdata[256];
    int bx = blockIdx.x;
    int l  = bx >> 8;
    int c  = bx & 255;
    const float* p = W + (size_t)l * HID * HID + (size_t)c * 16384;
    float s = 0.f;
    for (int i = threadIdx.x; i < 16384; i += 256) s += fabsf(p[i]);
    sdata[threadIdx.x] = s;
    __syncthreads();
    for (int off = 128; off > 0; off >>= 1) {
        if (threadIdx.x < off) sdata[threadIdx.x] += sdata[threadIdx.x + off];
        __syncthreads();
    }
    if (threadIdx.x == 0) partials[bx] = sdata[0];
}

__global__ void wabs_final(const float* __restrict__ partials, float* __restrict__ wscale) {
    __shared__ float sdata[256];
    int l = blockIdx.x;
    sdata[threadIdx.x] = partials[l * 256 + threadIdx.x];
    __syncthreads();
    for (int off = 128; off > 0; off >>= 1) {
        if (threadIdx.x < off) sdata[threadIdx.x] += sdata[threadIdx.x + off];
        __syncthreads();
    }
    if (threadIdx.x == 0)
        wscale[l] = sdata[0] / (float)(HID * HID) + EPSF;
}

__global__ void wquant(const float* __restrict__ W, const float* __restrict__ wscale,
                       uint32_t* __restrict__ tq) {
    size_t idx4 = (size_t)blockIdx.x * blockDim.x + threadIdx.x;
    int l = (int)((idx4 * 4) >> 22);
    float inv = 1.0f / wscale[l];
    float4 w = ((const float4*)W)[idx4];
    int q0 = (int)fmaxf(-1.f, fminf(1.f, rintf(w.x * inv)));
    int q1 = (int)fmaxf(-1.f, fminf(1.f, rintf(w.y * inv)));
    int q2 = (int)fmaxf(-1.f, fminf(1.f, rintf(w.z * inv)));
    int q3 = (int)fmaxf(-1.f, fminf(1.f, rintf(w.w * inv)));
    tq[idx4] = (uint32_t)(q0 & 255) | ((uint32_t)(q1 & 255) << 8) |
               ((uint32_t)(q2 & 255) << 16) | ((uint32_t)(q3 & 255) << 24);
}

__device__ inline uint32_t pack4q(float4 v, float s) {
    int q0 = (int)fmaxf(-128.f, fminf(127.f, rintf(v.x * s)));
    int q1 = (int)fmaxf(-128.f, fminf(127.f, rintf(v.y * s)));
    int q2 = (int)fmaxf(-128.f, fminf(127.f, rintf(v.z * s)));
    int q3 = (int)fmaxf(-128.f, fminf(127.f, rintf(v.w * s)));
    return (uint32_t)(q0 & 255) | ((uint32_t)(q1 & 255) << 8) |
           ((uint32_t)(q2 & 255) << 16) | ((uint32_t)(q3 & 255) << 24);
}

__global__ void actquant(const float* __restrict__ h, uint32_t* __restrict__ qx,
                         float* __restrict__ tokscale) {
    __shared__ float sdata[256];
    __shared__ float sbc;
    int row = blockIdx.x;
    const float4* hp = (const float4*)(h + (size_t)row * HID);
    float4 a = hp[threadIdx.x];
    float4 b = hp[threadIdx.x + 256];
    float m = fmaxf(fmaxf(fabsf(a.x), fabsf(a.y)), fmaxf(fabsf(a.z), fabsf(a.w)));
    m = fmaxf(m, fmaxf(fmaxf(fabsf(b.x), fabsf(b.y)), fmaxf(fabsf(b.z), fabsf(b.w))));
    sdata[threadIdx.x] = m;
    __syncthreads();
    for (int off = 128; off > 0; off >>= 1) {
        if (threadIdx.x < off)
            sdata[threadIdx.x] = fmaxf(sdata[threadIdx.x], sdata[threadIdx.x + off]);
        __syncthreads();
    }
    if (threadIdx.x == 0) {
        float amax = sdata[0] + EPSF;
        sbc = 127.0f / amax;
        tokscale[row] = amax / 127.0f;
    }
    __syncthreads();
    float s = sbc;
    uint32_t* op = qx + (size_t)row * (HID / 4);
    op[threadIdx.x]       = pack4q(a, s);
    op[threadIdx.x + 256] = pack4q(b, s);
}

__global__ void __launch_bounds__(256)
bitnet_gemm(const int8_t* __restrict__ A, const int8_t* __restrict__ T,
            const float* __restrict__ tokscale, const float* __restrict__ wscale_p,
            int layer, float* __restrict__ out) {
    const int lane   = threadIdx.x & 31;
    const int wave   = threadIdx.x >> 5;
    const int lanelo = lane & 15;
    const int lanehi = lane >> 4;
    const int m0 = blockIdx.x * 128 + (wave & 1) * 64;
    const int n0 = blockIdx.y * 256 + (wave >> 1) * 64;
    const int8_t* Bw = T + (size_t)layer * HID * HID;

    v8i acc[4][4];
    const v8i vz = {0, 0, 0, 0, 0, 0, 0, 0};
#pragma unroll
    for (int i = 0; i < 4; i++)
#pragma unroll
        for (int j = 0; j < 4; j++) acc[i][j] = vz;

    const int8_t* arow[4];
    const int8_t* brow[4];
#pragma unroll
    for (int i = 0; i < 4; i++) arow[i] = A  + (size_t)(m0 + 16 * i + lanelo) * HID;
#pragma unroll
    for (int j = 0; j < 4; j++) brow[j] = Bw + (size_t)(n0 + 16 * j + lanelo) * HID;

    for (int k0 = 0; k0 < HID; k0 += 64) {
        v8i bf[4];
#pragma unroll
        for (int j = 0; j < 4; j++) {
            const int4* p = (const int4*)(brow[j] + k0 + lanehi * 16);
            int4 x0 = p[0];
            int4 x1 = p[2];
            v8i f = {x0.x, x0.y, x0.z, x0.w, x1.x, x1.y, x1.z, x1.w};
            bf[j] = f;
        }
#pragma unroll
        for (int i = 0; i < 4; i++) {
            const int2* pa = (const int2*)(arow[i] + k0 + lanehi * 16);
            int2 a0 = pa[0];
            int2 a1 = pa[1];
            int2 a2 = pa[4];
            int2 a3 = pa[5];
            v8i af = {a0.x, a0.y, a1.x, a1.y, a2.x, a2.y, a3.x, a3.y};
#pragma unroll
            for (int j = 0; j < 4; j++)
                acc[i][j] = __builtin_amdgcn_wmma_i32_16x16x64_iu8(
                     true, af,  true, bf[j], acc[i][j],
                     false,  false);
        }
    }

    const float wsc = wscale_p[layer];
#pragma unroll
    for (int i = 0; i < 4; i++) {
        float rs[8];
#pragma unroll
        for (int g = 0; g < 8; g++)
            rs[g] = wsc * tokscale[m0 + 16 * i + 8 * lanehi + g];
#pragma unroll
        for (int j = 0; j < 4; j++) {
            float* op = out + (size_t)(m0 + 16 * i + 8 * lanehi) * HID + n0 + 16 * j + lanelo;
#pragma unroll
            for (int g = 0; g < 8; g++)
                op[(size_t)g * HID] = (float)acc[i][j][g] * rs[g];
        }
    }
}

extern "C" void kernel_launch(void* const* d_in, const int* in_sizes, int n_in,
                              void* d_out, int out_size, void* d_ws, size_t ws_size,
                              hipStream_t stream) {
    const float* x = (const float*)d_in[0];
    const float* W = (const float*)d_in[1];
    float* out = (float*)d_out;
    char* ws = (char*)d_ws;

    int8_t* qx = (int8_t*)ws;
    int8_t* tq = (int8_t*)(ws + (size_t)NTOK * HID);
    float* tokscale = (float*)(ws + (size_t)NTOK * HID + (size_t)NLAYER * HID * HID);
    float* wscale   = tokscale + NTOK;
    float* partials = wscale + NLAYER;

    wabs_partial<<<NLAYER * 256, 256, 0, stream>>>(W, partials);
    wabs_final<<<NLAYER, 256, 0, stream>>>(partials, wscale);
    wquant<<<(NLAYER * HID * HID / 4) / 256, 256, 0, stream>>>(W, wscale, (uint32_t*)tq);

    dim3 ggrid(NTOK / 128, HID / 256);
    for (int l = 0; l < NLAYER; ++l) {
        const float* hin = (l == 0) ? x : out;
        actquant<<<NTOK, 256, 0, stream>>>(hin, (uint32_t*)qx, tokscale);
        bitnet_gemm<<<ggrid, 256, 0, stream>>>(qx, tq, tokscale, wscale, l, out);
    }
}
